// MOEMamba_59528246723226
// MI455X (gfx1250) — matrix operand order repaired and checked by reading, not compiled, not run on hardware
//
#include <hip/hip_runtime.h>
#include <hip/hip_bf16.h>
#include <stddef.h>

// ---------------- model constants ----------------
#define NB_   8
#define DIM_  256
#define DI_   512
#define DCONV_ 4
#define NS_   16
#define DTR_  16
#define NE_   4
#define INNER_ 512
#define LTOK  8192
#define HOR_  32

typedef __attribute__((ext_vector_type(16))) __bf16 v16bf;
typedef __attribute__((ext_vector_type(8)))  __bf16 v8bf;
typedef __attribute__((ext_vector_type(8)))  float  v8f;

__device__ __forceinline__ __bf16 f2bf(float f) {
    unsigned u = __builtin_bit_cast(unsigned, f);
    unsigned r = (u + 0x7FFFu + ((u >> 16) & 1u)) >> 16;
    return __builtin_bit_cast(__bf16, (unsigned short)r);
}
__device__ __forceinline__ float sigm_f(float x) { return 1.f / (1.f + __expf(-x)); }
__device__ __forceinline__ float silu_f(float x) { return x * sigm_f(x); }

// =====================================================================
// one-time fp32 -> bf16 conversion, 8 elems/thread (n % 8 == 0)
// =====================================================================
__global__ __launch_bounds__(256) void cvt8_k(const float* __restrict__ in,
                                              __bf16* __restrict__ out, int n8)
{
    int idx = blockIdx.x * 256 + threadIdx.x;
    if (idx >= n8) return;
    const float4* p = (const float4*)(in + (size_t)idx * 8);
    float4 a = p[0], b = p[1];
    v8bf st;
    st[0]=f2bf(a.x); st[1]=f2bf(a.y); st[2]=f2bf(a.z); st[3]=f2bf(a.w);
    st[4]=f2bf(b.x); st[5]=f2bf(b.y); st[6]=f2bf(b.z); st[7]=f2bf(b.w);
    *(v8bf*)(out + (size_t)idx * 8) = st;
}

// =====================================================================
// WMMA GEMM:  C[M,N] = A[M,K] * B[N,K]^T   (A,B bf16, f32 accum)
// Block tile 64x128, BK=32, 256 threads (8 waves 2x4), wave tile 32x32.
// REQUIRES: K % 32 == 0, M % 64 == 0. Out-of-range N rows clamped (their
// products only reach epilogue-masked columns). Software-pipelined.
// MODE: 0 store, 1 silu, 2 mul-by-E, 3 rowscale(+beta) accumulate, 4 add-E
// WB:   0 fp32 only, 1 fp32 + bf16 copy, 2 bf16 only
// =====================================================================
template<int MODE, int WB>
__global__ __launch_bounds__(256) void gemm_bf16w(
    const __bf16* __restrict__ A, const __bf16* __restrict__ B,
    float* __restrict__ C, __bf16* __restrict__ Cb, int M, int N, int K,
    const float* __restrict__ E, const float* __restrict__ RS, int beta)
{
    __shared__ __align__(16) __bf16 As[64 * 48];
    __shared__ __align__(16) __bf16 Bs[128 * 48];

    const int tid   = threadIdx.x;
    const int mBase = blockIdx.y * 64;
    const int nBase = blockIdx.x * 128;
    const int lane  = tid & 31;
    const int wave  = tid >> 5;
    const int wm    = wave >> 2;        // 0..1
    const int wn    = wave & 3;         // 0..3
    const int lr    = lane & 15;
    const int lh    = lane >> 4;        // lane-half select

    // staging coordinates (branch-free; N rows clamped)
    const int ar  = tid >> 2;
    const int ac  = (tid & 3) << 3;
    const int br  = tid >> 1;
    const int bc  = (tid & 1) << 4;
    const int bgr = min(nBase + br, N - 1);
    const __bf16* __restrict__ aptr = A + (size_t)(mBase + ar) * K + ac;
    const __bf16* __restrict__ bptr = B + (size_t)bgr * K + bc;

    const v8f zacc = {0.f,0.f,0.f,0.f,0.f,0.f,0.f,0.f};
    v8f acc[2][2];
    acc[0][0] = zacc; acc[0][1] = zacc; acc[1][0] = zacc; acc[1][1] = zacc;

    // prologue load (tile 0)
    v8bf aS  = *(const v8bf*)(aptr);
    v8bf bS0 = *(const v8bf*)(bptr);
    v8bf bS1 = *(const v8bf*)(bptr + 8);

    const int kTiles = K >> 5;
    for (int kk = 0; kk < kTiles; ++kk) {
        *(v8bf*)(&As[ar * 48 + ac])     = aS;
        *(v8bf*)(&Bs[br * 48 + bc])     = bS0;
        *(v8bf*)(&Bs[br * 48 + bc + 8]) = bS1;
        __syncthreads();

        // ---- issue global loads for next tile (overlap with WMMA below)
        if (kk + 1 < kTiles) {
            const int k0 = (kk + 1) << 5;
            aS  = *(const v8bf*)(aptr + k0);
            bS0 = *(const v8bf*)(bptr + k0);
            bS1 = *(const v8bf*)(bptr + k0 + 8);
        }

        // ---- B fragments (lane lr = col N; K interleave {0-7,16-23}/{8-15,24-31})
        v16bf bfr[2];
        #pragma unroll
        for (int nt = 0; nt < 2; ++nt) {
            const __bf16* bb = &Bs[(wn * 32 + nt * 16 + lr) * 48 + lh * 8];
            v8bf blo = *(const v8bf*)bb;
            v8bf bhi = *(const v8bf*)(bb + 16);
            bfr[nt] = __builtin_shufflevector(blo, bhi,
                0,1,2,3,4,5,6,7,8,9,10,11,12,13,14,15);
        }
        // ---- A fragments (lane lr = row M; K interleave {0-7,16-23}/{8-15,24-31})
        #pragma unroll
        for (int mt = 0; mt < 2; ++mt) {
            const __bf16* ab = &As[(wm * 32 + mt * 16 + lr) * 48 + lh * 8];
            v8bf alo = *(const v8bf*)ab;
            v8bf ahi = *(const v8bf*)(ab + 16);
            v16bf afr = __builtin_shufflevector(alo, ahi,
                0,1,2,3,4,5,6,7,8,9,10,11,12,13,14,15);
            #pragma unroll
            for (int nt = 0; nt < 2; ++nt) {
                acc[mt][nt] = __builtin_amdgcn_wmma_f32_16x16x32_bf16(
                    false, afr, false, bfr[nt], (short)0, acc[mt][nt], false, false);
            }
        }
        __syncthreads();
    }

    // ---- epilogue: VGPR r -> row r + 8*lh, col = lr
    #pragma unroll
    for (int mt = 0; mt < 2; ++mt) {
        #pragma unroll
        for (int nt = 0; nt < 2; ++nt) {
            #pragma unroll
            for (int r = 0; r < 8; ++r) {
                int row = mBase + wm * 32 + mt * 16 + lh * 8 + r;
                int col = nBase + wn * 32 + nt * 16 + lr;
                if (col < N) {
                    float v = acc[mt][nt][r];
                    size_t o = (size_t)row * N + col;
                    float f;
                    if (MODE == 0)      f = v;
                    else if (MODE == 1) f = silu_f(v);
                    else if (MODE == 2) f = v * E[o];
                    else if (MODE == 3) {
                        float s = RS[(size_t)row * 4];
                        f = beta ? (C[o] + s * v) : (s * v);
                    }
                    else                f = v + E[o];   // MODE 4
                    if (WB != 2) C[o]  = f;
                    if (WB >= 1) Cb[o] = f2bf(f);
                }
            }
        }
    }
}

// =====================================================================
// transpose (DIM,L) -> (L,DIM), emit bf16 (feeds GEMM A only)
// =====================================================================
__global__ __launch_bounds__(256) void transpose_x(const float* __restrict__ x,
                                                   __bf16* __restrict__ ht)
{
    int idx = blockIdx.x * 256 + threadIdx.x;
    if (idx >= DIM_ * LTOK) return;
    int d = idx >> 13;          // /L
    int l = idx & (LTOK - 1);
    ht[(size_t)l * DIM_ + d] = f2bf(x[idx]);
}

// =====================================================================
// rmsnorm over 256-dim rows; one wave per row.
// ADD=1 fuses residual add. WF=1 also writes fp32. bf16 always written.
// =====================================================================
template<int ADD, int WF>
__global__ __launch_bounds__(256) void rmsnorm_k(const float* __restrict__ in,
                                                 const float* __restrict__ w,
                                                 float* __restrict__ outf,
                                                 __bf16* __restrict__ outb,
                                                 const float* __restrict__ addin)
{
    int wave = threadIdx.x >> 5, lane = threadIdx.x & 31;
    int row  = blockIdx.x * 8 + wave;
    const float* ip = in + (size_t)row * DIM_;
    float v[8]; float ss = 0.f;
    #pragma unroll
    for (int i = 0; i < 8; ++i) { v[i] = ip[lane + i * 32]; ss += v[i] * v[i]; }
    ss += __shfl_xor(ss, 16); ss += __shfl_xor(ss, 8);
    ss += __shfl_xor(ss, 4);  ss += __shfl_xor(ss, 2); ss += __shfl_xor(ss, 1);
    float sc = rsqrtf(ss * (1.f / DIM_) + 1e-3f);
    #pragma unroll
    for (int i = 0; i < 8; ++i) {
        int c = lane + i * 32;
        float o = w[c] * v[i] * sc;
        if (ADD) o += addin[(size_t)row * DIM_ + c];
        if (WF)  outf[(size_t)row * DIM_ + c] = o;
        outb[(size_t)row * DIM_ + c] = f2bf(o);
    }
}

// =====================================================================
// causal depthwise conv (k=4) + bias + silu -> xc fp32 (scan) + bf16 (GEMM)
// =====================================================================
__global__ __launch_bounds__(256) void conv_silu_k(const float* __restrict__ xr,
                                                   const float* __restrict__ cw,
                                                   const float* __restrict__ cb,
                                                   float* __restrict__ xc,
                                                   __bf16* __restrict__ xcb)
{
    int idx = blockIdx.x * 256 + threadIdx.x;
    if (idx >= LTOK * DI_) return;
    int d = idx & (DI_ - 1);
    int l = idx >> 9;
    float acc = cb[d];
    #pragma unroll
    for (int j = 0; j < DCONV_; ++j) {
        int ll = l + j - (DCONV_ - 1);
        if (ll >= 0) acc += cw[d * DCONV_ + j] * xr[(size_t)ll * (2 * DI_) + d];
    }
    float s = silu_f(acc);
    xc[(size_t)l * DI_ + d]  = s;
    xcb[(size_t)l * DI_ + d] = f2bf(s);
}

// =====================================================================
// delta = softplus(dr @ dtw^T + dtb), dr = xdbl[:, :16]
// =====================================================================
__global__ __launch_bounds__(256) void delta_k(const float* __restrict__ xdbl,
                                               const float* __restrict__ dtw,
                                               const float* __restrict__ dtb,
                                               float* __restrict__ dlt)
{
    int idx = blockIdx.x * 256 + threadIdx.x;
    if (idx >= LTOK * DI_) return;
    int d = idx & (DI_ - 1);
    int l = idx >> 9;
    float acc = dtb[d];
    #pragma unroll
    for (int r = 0; r < DTR_; ++r)
        acc += xdbl[(size_t)l * 48 + r] * dtw[d * DTR_ + r];
    float sp = (acc > 20.f) ? acc : log1pf(__expf(acc));
    dlt[(size_t)l * DI_ + d] = sp;
}

// =====================================================================
// selective scan: lane = (d x 16 states), 16-lane shuffle reductions.
// y = (scan + xc*D) * silu(res)  -> bf16 (feeds out_proj GEMM only)
// =====================================================================
__global__ __launch_bounds__(256) void scan_k(const float* __restrict__ dlt,
                                              const float* __restrict__ xdbl,
                                              const float* __restrict__ xc,
                                              const float* __restrict__ xr,
                                              const float* __restrict__ alog,
                                              const float* __restrict__ dpar,
                                              __bf16* __restrict__ yb)
{
    int n    = threadIdx.x & 15;
    int dloc = threadIdx.x >> 4;
    int d    = blockIdx.x * 16 + dloc;
    float An = -__expf(alog[d * NS_ + n]);
    float Dd = dpar[d];
    float carry = 0.f;
    for (int l = 0; l < LTOK; ++l) {
        float de  = dlt[(size_t)l * DI_ + d];
        float xcv = xc[(size_t)l * DI_ + d];
        float Bv  = xdbl[(size_t)l * 48 + DTR_ + n];
        float Cv  = xdbl[(size_t)l * 48 + DTR_ + NS_ + n];
        carry = __expf(de * An) * carry + de * Bv * xcv;
        float part = carry * Cv;
        part += __shfl_xor(part, 1);
        part += __shfl_xor(part, 2);
        part += __shfl_xor(part, 4);
        part += __shfl_xor(part, 8);
        if (n == 0) {
            float rv = xr[(size_t)l * (2 * DI_) + DI_ + d];
            yb[(size_t)l * DI_ + d] = f2bf((part + xcv * Dd) * silu_f(rv));
        }
    }
}

// =====================================================================
// gating: scores, top-2, softmax weights -> we (L,4), zeros elsewhere
// =====================================================================
__global__ __launch_bounds__(256) void gate_k(const float* __restrict__ h,
                                              const float* __restrict__ gw,
                                              float* __restrict__ we)
{
    int wave = threadIdx.x >> 5, lane = threadIdx.x & 31;
    int t = blockIdx.x * 8 + wave;
    float s0 = 0.f, s1 = 0.f, s2 = 0.f, s3 = 0.f;
    for (int c = lane; c < DIM_; c += 32) {
        float hv = h[(size_t)t * DIM_ + c];
        s0 += hv * gw[0 * DIM_ + c];
        s1 += hv * gw[1 * DIM_ + c];
        s2 += hv * gw[2 * DIM_ + c];
        s3 += hv * gw[3 * DIM_ + c];
    }
    #pragma unroll
    for (int off = 16; off >= 1; off >>= 1) {
        s0 += __shfl_xor(s0, off); s1 += __shfl_xor(s1, off);
        s2 += __shfl_xor(s2, off); s3 += __shfl_xor(s3, off);
    }
    if (lane == 0) {
        float s[4] = {s0, s1, s2, s3};
        int i1 = 0;
        for (int e = 1; e < 4; ++e) if (s[e] > s[i1]) i1 = e;
        int i2 = -1;
        for (int e = 0; e < 4; ++e) if (e != i1 && (i2 < 0 || s[e] > s[i2])) i2 = e;
        float w1 = 1.f / (1.f + __expf(s[i2] - s[i1]));
        we[(size_t)t * 4 + 0] = 0.f; we[(size_t)t * 4 + 1] = 0.f;
        we[(size_t)t * 4 + 2] = 0.f; we[(size_t)t * 4 + 3] = 0.f;
        we[(size_t)t * 4 + i1] = w1;
        we[(size_t)t * 4 + i2] = 1.f - w1;
    }
}

// =====================================================================
// final sigmoid + transpose (L,32) -> (32,L)
// =====================================================================
__global__ __launch_bounds__(256) void sig_tr_k(const float* __restrict__ t3,
                                                float* __restrict__ out)
{
    int idx = blockIdx.x * 256 + threadIdx.x;
    if (idx >= HOR_ * LTOK) return;
    int o = idx >> 13;
    int l = idx & (LTOK - 1);
    out[idx] = sigm_f(t3[(size_t)l * HOR_ + o]);
}

// =====================================================================
// host launcher
// =====================================================================
static inline dim3 ggrid(int M, int N) { return dim3((N + 127) / 128, (M + 63) / 64); }

extern "C" void kernel_launch(void* const* d_in, const int* in_sizes, int n_in,
                              void* d_out, int out_size, void* d_ws, size_t ws_size,
                              hipStream_t stream) {
    (void)in_sizes; (void)n_in; (void)out_size; (void)ws_size;
    const float* x       = (const float*)d_in[0];
    const float* lin_w   = (const float*)d_in[1];
    const float* ipw_a   = (const float*)d_in[2];
    const float* cw_a    = (const float*)d_in[3];
    const float* cb_a    = (const float*)d_in[4];
    const float* xpw_a   = (const float*)d_in[5];
    const float* dtw_a   = (const float*)d_in[6];
    const float* dtb_a   = (const float*)d_in[7];
    const float* alog_a  = (const float*)d_in[8];
    const float* dpar_a  = (const float*)d_in[9];
    const float* opw_a   = (const float*)d_in[10];
    const float* gw_a    = (const float*)d_in[11];
    const float* gp_a    = (const float*)d_in[12];
    const float* up_a    = (const float*)d_in[13];
    const float* dp_a    = (const float*)d_in[14];
    const float* rms_a   = (const float*)d_in[15];
    const float* rms_f   = (const float*)d_in[16];
    const float* hw1     = (const float*)d_in[17];
    const float* hw2     = (const float*)d_in[18];
    float* out = (float*)d_out;

    const size_t LL = LTOK;
    float* W    = (float*)d_ws;
    // ---- fp32 region
    float* h    = W + 0;            // (L,256)
    float* h2   = W + 256 * LL;     // (L,256)
    float* xr   = W + 512 * LL;     // (L,1024)
    float* xc   = W + 1536 * LL;    // (L,512)   also G
    float* xdbl = W + 2048 * LL;    // (L,48)    also t3
    float* dlt  = W + 2096 * LL;    // (L,512)
    float* moe  = W + 2608 * LL;    // (L,256)
    float* we   = W + 2864 * LL;    // (L,4)
    // ---- bf16 region (activation mirrors + converted weights)
    __bf16* H    = (__bf16*)(W + 2868 * LL);
    __bf16* hnb  = H + 0;           // (L,256)  ht / hn / t1
    __bf16* xcb  = H + 256 * LL;    // (L,512)  xc / t2
    __bf16* ybb  = H + 768 * LL;    // (L,512)
    __bf16* h2b  = H + 1280 * LL;   // (L,256)
    __bf16* GUb  = H + 1536 * LL;   // (L,512)
    __bf16* hb   = H + 2048 * LL;   // (L,256)
    __bf16* wp   = H + 2304 * LL;
    __bf16* linb = wp; wp += (size_t)DIM_ * DIM_;
    __bf16* ipwb = wp; wp += (size_t)NB_ * 2 * DI_ * DIM_;
    __bf16* xpwb = wp; wp += (size_t)NB_ * (DTR_ + 2 * NS_) * DI_;
    __bf16* opwb = wp; wp += (size_t)NB_ * DIM_ * DI_;
    __bf16* gpb  = wp; wp += (size_t)NB_ * NE_ * INNER_ * DIM_;
    __bf16* upb  = wp; wp += (size_t)NB_ * NE_ * INNER_ * DIM_;
    __bf16* dpb  = wp; wp += (size_t)NB_ * NE_ * DIM_ * INNER_;
    __bf16* hw1b = wp; wp += (size_t)DIM_ * DIM_;
    __bf16* hw2b = wp; wp += (size_t)HOR_ * DIM_;

    dim3 blk(256);
    const int L = LTOK;
    auto cvt = [&](const float* src, __bf16* dst, size_t n) {
        int n8 = (int)(n / 8);
        cvt8_k<<<(n8 + 255) / 256, blk, 0, stream>>>(src, dst, n8);
    };
    // ---- one-time weight conversions (bandwidth-trivial)
    cvt(lin_w, linb, (size_t)DIM_ * DIM_);
    cvt(ipw_a, ipwb, (size_t)NB_ * 2 * DI_ * DIM_);
    cvt(xpw_a, xpwb, (size_t)NB_ * (DTR_ + 2 * NS_) * DI_);
    cvt(opw_a, opwb, (size_t)NB_ * DIM_ * DI_);
    cvt(gp_a,  gpb,  (size_t)NB_ * NE_ * INNER_ * DIM_);
    cvt(up_a,  upb,  (size_t)NB_ * NE_ * INNER_ * DIM_);
    cvt(dp_a,  dpb,  (size_t)NB_ * NE_ * DIM_ * INNER_);
    cvt(hw1,   hw1b, (size_t)DIM_ * DIM_);
    cvt(hw2,   hw2b, (size_t)HOR_ * DIM_);

    // h = x.T @ lin_w.T
    transpose_x<<<(DIM_ * L + 255) / 256, blk, 0, stream>>>(x, hnb);
    gemm_bf16w<0,0><<<ggrid(L, DIM_), blk, 0, stream>>>(hnb, linb, h, nullptr,
        L, DIM_, DIM_, nullptr, nullptr, 0);

    for (int i = 0; i < NB_; ++i) {
        const __bf16* ipw  = ipwb + (size_t)i * 2 * DI_ * DIM_;
        const float*  cw   = cw_a  + (size_t)i * DI_ * DCONV_;
        const float*  cb   = cb_a  + (size_t)i * DI_;
        const __bf16* xpw  = xpwb + (size_t)i * (DTR_ + 2 * NS_) * DI_;
        const float*  dtw  = dtw_a + (size_t)i * DI_ * DTR_;
        const float*  dtb  = dtb_a + (size_t)i * DI_;
        const float*  alog = alog_a + (size_t)i * DI_ * NS_;
        const float*  dpar = dpar_a + (size_t)i * DI_;
        const __bf16* opw  = opwb + (size_t)i * DIM_ * DI_;
        const float*  gw   = gw_a  + (size_t)i * NE_ * DIM_;

        // ---- mamba ----
        rmsnorm_k<0,0><<<L / 8, blk, 0, stream>>>(h, rms_a, nullptr, hnb, nullptr);
        gemm_bf16w<0,0><<<ggrid(L, 2 * DI_), blk, 0, stream>>>(hnb, ipw, xr, nullptr,
            L, 2 * DI_, DIM_, nullptr, nullptr, 0);
        conv_silu_k<<<(L * DI_ + 255) / 256, blk, 0, stream>>>(xr, cw, cb, xc, xcb);
        gemm_bf16w<0,0><<<ggrid(L, 48), blk, 0, stream>>>(xcb, xpw, xdbl, nullptr,
            L, 48, DI_, nullptr, nullptr, 0);
        delta_k<<<(L * DI_ + 255) / 256, blk, 0, stream>>>(xdbl, dtw, dtb, dlt);
        scan_k<<<DI_ / 16, blk, 0, stream>>>(dlt, xdbl, xc, xr, alog, dpar, ybb);
        // h2 = y @ opw^T + h   (residual); bf16 mirror for MoE GEMMs
        gemm_bf16w<4,1><<<ggrid(L, DIM_), blk, 0, stream>>>(ybb, opw, h2, h2b,
            L, DIM_, DI_, h, nullptr, 0);

        // ---- moe ----
        gate_k<<<L / 8, blk, 0, stream>>>(h2, gw, we);
        float* G = xc;     // xc fp32 dead after scan
        for (int e = 0; e < NE_; ++e) {
            const __bf16* gp = gpb + ((size_t)i * NE_ + e) * INNER_ * DIM_;
            const __bf16* up = upb + ((size_t)i * NE_ + e) * INNER_ * DIM_;
            const __bf16* dp = dpb + ((size_t)i * NE_ + e) * DIM_ * INNER_;
            gemm_bf16w<1,0><<<ggrid(L, INNER_), blk, 0, stream>>>(h2b, gp, G, nullptr,
                L, INNER_, DIM_, nullptr, nullptr, 0);
            gemm_bf16w<2,2><<<ggrid(L, INNER_), blk, 0, stream>>>(h2b, up, nullptr, GUb,
                L, INNER_, DIM_, G, nullptr, 0);
            gemm_bf16w<3,0><<<ggrid(L, DIM_), blk, 0, stream>>>(GUb, dp, moe, nullptr,
                L, DIM_, INNER_, nullptr, we + e, (e > 0) ? 1 : 0);
        }
        // h = rmsnorm(moe) + h2 (fp32 for next layer, bf16 for head GEMM)
        rmsnorm_k<1,1><<<L / 8, blk, 0, stream>>>(moe, rms_f, h, hb, h2);
    }

    // ---- head ----
    __bf16* t1b = hnb;
    __bf16* t2b = xcb;
    float*  t3  = xdbl;
    gemm_bf16w<0,2><<<ggrid(L, DIM_), blk, 0, stream>>>(hb, linb, nullptr, t1b,
        L, DIM_, DIM_, nullptr, nullptr, 0);
    gemm_bf16w<1,2><<<ggrid(L, DIM_), blk, 0, stream>>>(t1b, hw1b, nullptr, t2b,
        L, DIM_, DIM_, nullptr, nullptr, 0);
    gemm_bf16w<0,0><<<ggrid(L, HOR_), blk, 0, stream>>>(t2b, hw2b, t3, nullptr,
        L, HOR_, DIM_, nullptr, nullptr, 0);
    sig_tr_k<<<(HOR_ * L + 255) / 256, blk, 0, stream>>>(t3, out);
}
